// DynamicEdgeWeights_4002909520446
// MI455X (gfx1250) — matrix operand order repaired and checked by reading, not compiled, not run on hardware
//
#include <hip/hip_runtime.h>
#include <hip/hip_bf16.h>

typedef _Float16 v16h __attribute__((ext_vector_type(16)));
typedef _Float16 h8   __attribute__((ext_vector_type(8)));
typedef float    v8f  __attribute__((ext_vector_type(8)));

#define Bsz 4
#define Ssz 512
#define Dsz 128
#define Hsz 8

// ---------------------------------------------------------------------------
// Kernel 1: projections.  aP[row][e] = x[row]·w1[:,e] + b1[e] (f16),
//                         cP[row][e] = x[row]·w1[D:,e]        (f16)
// 2048 rows x 128 cols, K=128 -> trivial (0.13 GFLOP), feeds L2-resident f16.
// ---------------------------------------------------------------------------
__global__ void __launch_bounds__(128)
proj_kernel(const float* __restrict__ x,
            const float* __restrict__ w1,
            const float* __restrict__ b1,
            _Float16* __restrict__ aP,
            _Float16* __restrict__ cP) {
  const int row = blockIdx.x;      // 0..B*S-1
  const int e   = threadIdx.x;     // 0..127  (coalesced over w1 columns)
  const float* xr = x + (size_t)row * Dsz;
  float acc_a = 0.f, acc_c = 0.f;
#pragma unroll 8
  for (int k = 0; k < Dsz; ++k) {
    const float xv = xr[k];
    acc_a = fmaf(xv, w1[(size_t)k * Dsz + e], acc_a);
    acc_c = fmaf(xv, w1[(size_t)(k + Dsz) * Dsz + e], acc_c);
  }
  aP[(size_t)row * Dsz + e] = (_Float16)(acc_a + b1[e]);  // fold b1 into a'
  cP[(size_t)row * Dsz + e] = (_Float16)acc_c;
}

// ---------------------------------------------------------------------------
// Kernel 2: fused  sigmoid( relu(a'_i + c_j) @ w2 + b2 )  ->  out[b,h,i,j]
// One block per (b,i); 8 waves; each wave does 4 j-tiles of 16.
// Per tile: build A (16x32 f16) in registers, 4x v_wmma_f32_16x16x32_f16.
// ---------------------------------------------------------------------------
__global__ void __launch_bounds__(256)
edge_kernel(const _Float16* __restrict__ aP,
            const _Float16* __restrict__ cP,
            const float* __restrict__ w2,
            const float* __restrict__ b2,
            float* __restrict__ out) {
  const int bi   = blockIdx.x;            // b*S + i
  const int b    = bi / Ssz;
  const int i    = bi % Ssz;
  const int lane = threadIdx.x & 31;      // wave32
  const int wave = threadIdx.x >> 5;

  const int n    = lane & 15;             // output h column (valid if < 8)
  const int hiH  = lane >> 4;             // lane half-select
  const int koff = hiH * 8;               // A-frag K sub-offset per layout

  // --- B fragments: w2 [128][8] zero-padded to 16 cols, one v16h per K-chunk.
  // Lane l: N = l%16, K = chunk*32 + (l>=16?8:0) + t + (t>=8?8:0).  Loop-invariant.
  v16h Bfrag[4];
#pragma unroll
  for (int q = 0; q < 4; ++q) {
    const int kb = q * 32 + hiH * 8;
#pragma unroll
    for (int t = 0; t < 16; ++t) {
      const float wv = (n < Hsz) ? w2[(size_t)(kb + t + (t >= 8 ? 8 : 0)) * Hsz + n] : 0.f;
      Bfrag[q][t] = (_Float16)wv;
    }
  }
  const float bias2 = (n < Hsz) ? b2[n] : 0.f;

  // --- a' runs for this (b,i): per chunk q, two aligned 8-half runs.
  const _Float16* arow = aP + (size_t)bi * Dsz;
  h8 aruns[4][2];
#pragma unroll
  for (int q = 0; q < 4; ++q) {
    aruns[q][0] = *(const h8*)(arow + q * 32 + koff);
    aruns[q][1] = *(const h8*)(arow + q * 32 + koff + 16);
  }

  const int jlocal = lane & 15;           // A-matrix row M = local j

  for (int jt = wave; jt < Ssz / 16; jt += 8) {
    const int j0 = jt * 16;
    const _Float16* crow = cP + (size_t)(b * Ssz + j0 + jlocal) * Dsz;

    // Prefetch next j-tile's c rows while this tile computes.
    if (jt + 8 < Ssz / 16)
      __builtin_prefetch(cP + (size_t)(b * Ssz + j0 + 128 + jlocal) * Dsz, 0, 1);

    v8f acc = {};
#pragma unroll
    for (int q = 0; q < 4; ++q) {
      const h8 c0 = *(const h8*)(crow + q * 32 + koff);
      const h8 c1 = *(const h8*)(crow + q * 32 + koff + 16);
      v16h A;
#pragma unroll
      for (int t = 0; t < 8; ++t) {         // relu(a'+c) -> v_pk f16 ops
        _Float16 s0 = aruns[q][0][t] + c0[t];
        _Float16 s1 = aruns[q][1][t] + c1[t];
        A[t]     = (s0 > (_Float16)0) ? s0 : (_Float16)0;
        A[t + 8] = (s1 > (_Float16)0) ? s1 : (_Float16)0;
      }
      acc = __builtin_amdgcn_wmma_f32_16x16x32_f16(
          /*neg_a=*/false, A, /*neg_b=*/false, Bfrag[q],
          /*c_mod=*/(short)0, acc, /*reuse_a=*/false, /*reuse_b=*/false);
    }

    // Epilogue: sigmoid(acc + b2); lane holds rows j0 + r + 8*hiH, col n.
    if (n < Hsz) {
      float vals[8];
#pragma unroll
      for (int r = 0; r < 8; ++r) {
        const float v = acc[r] + bias2;
        vals[r] = 1.0f / (1.0f + __expf(-v));
      }
      float* op = out + ((size_t)(b * Hsz + n) * Ssz + i) * Ssz + j0 + hiH * 8;
      *(float4*)(op)     = make_float4(vals[0], vals[1], vals[2], vals[3]);
      *(float4*)(op + 4) = make_float4(vals[4], vals[5], vals[6], vals[7]);
    }
  }
}

// ---------------------------------------------------------------------------
extern "C" void kernel_launch(void* const* d_in, const int* in_sizes, int n_in,
                              void* d_out, int out_size, void* d_ws, size_t ws_size,
                              hipStream_t stream) {
  const float* x  = (const float*)d_in[0];   // [B,S,D]
  const float* w1 = (const float*)d_in[1];   // [2D,D]
  const float* b1 = (const float*)d_in[2];   // [D]
  const float* w2 = (const float*)d_in[3];   // [D,H]
  const float* b2 = (const float*)d_in[4];   // [H]
  float* out = (float*)d_out;                // [B,H,S,S]

  _Float16* aP = (_Float16*)d_ws;                       // 512 KB
  _Float16* cP = aP + (size_t)Bsz * Ssz * Dsz;          // 512 KB

  proj_kernel<<<Bsz * Ssz, Dsz, 0, stream>>>(x, w1, b1, aP, cP);
  edge_kernel<<<Bsz * Ssz, 256, 0, stream>>>(aP, cP, w2, b2, out);
}
